// LineageLinkPredictionGNN_21028159881504
// MI455X (gfx1250) — matrix operand order repaired and checked by reading, not compiled, not run on hardware
//
#include <hip/hip_runtime.h>
#include <hip/hip_bf16.h>

// ---------------------------------------------------------------------------
// MI455X (gfx1250, wave32) fused GCN link-prediction pipeline.
// Memory-bound workload (~2.6 GB moved vs ~30 GFLOP): all dense math uses
// v_wmma_f32_16x16x32_bf16 (f32 accumulate), activations are carried in bf16
// to halve gather traffic, scatter-add uses f32 global atomics per reference
// semantics. N (50000) and E (800000) are multiples of 16 -> exact tiling.
// ---------------------------------------------------------------------------

typedef __attribute__((ext_vector_type(16))) __bf16        v16bf;
typedef __attribute__((ext_vector_type(8)))  float         v8f;
typedef __attribute__((ext_vector_type(4)))  unsigned int  u32x4;
typedef __attribute__((ext_vector_type(2)))  unsigned int  u32x2;
typedef __attribute__((ext_vector_type(4)))  float         f32x4;

union AF { v16bf v; u32x4 q[2]; };

__device__ __forceinline__ unsigned short f2bf(float f) {
    unsigned u = __builtin_bit_cast(unsigned, f);
    u += 0x7FFFu + ((u >> 16) & 1u);            // round-to-nearest-even
    return (unsigned short)(u >> 16);
}

// ---------------- degree / normalization ----------------
__global__ void k_init_deg(float* deg, int n) {
    int i = blockIdx.x * 256 + threadIdx.x;
    if (i < n) deg[i] = 1.0f;                   // self-loop contribution
}
__global__ void k_count_deg(const int* ei, float* deg, int E) {
    int e = blockIdx.x * 256 + threadIdx.x;
    if (e < E) atomicAdd(&deg[ei[E + e]], 1.0f);
}
__global__ void k_rsqrt(float* deg, int n) {
    int i = blockIdx.x * 256 + threadIdx.x;
    if (i < n) deg[i] = rsqrtf(deg[i]);         // deg >= 1 always
}

// ---------------- f32 -> bf16 bulk convert (4 elems/thread) ----------------
__global__ void k_f32_to_bf16_x4(const float* __restrict__ src,
                                 unsigned short* __restrict__ dst, int n4) {
    int i = blockIdx.x * 256 + threadIdx.x;
    if (i >= n4) return;
    f32x4 v = ((const f32x4*)src)[i];
    u32x2 w;
    w.x = (unsigned)f2bf(v.x) | ((unsigned)f2bf(v.y) << 16);
    w.y = (unsigned)f2bf(v.z) | ((unsigned)f2bf(v.w) << 16);
    ((u32x2*)dst)[i] = w;
}

// ---------------- pack weight [K,Nc] f32 -> WMMA B-fragment bf16 layout ----
// frag id = ktile*(Nc/16)+ntile; per lane 16 contiguous halves:
//   element i -> K = ktile*32 + (lane>>4)*8 + (i&7) + (i>>3)*16, col = ntile*16 + (lane&15)
__global__ void k_pack_w(const float* __restrict__ W,
                         unsigned short* __restrict__ Wp, int K, int Nc) {
    int t = blockIdx.x * 256 + threadIdx.x;
    if (t >= K * Nc) return;
    int NT     = Nc >> 4;
    int frag   = t >> 9;         // 512 halves per 32x16 fragment
    int within = t & 511;
    int lane   = within >> 4;
    int i      = within & 15;
    int kt = frag / NT, nt = frag % NT;
    int k = kt * 32 + (lane >> 4) * 8 + (i & 7) + (i >> 3) * 16;
    int n = nt * 16 + (lane & 15);
    Wp[t] = f2bf(W[k * Nc + n]);
}

// ---------------- zero ----------------
__global__ void k_zero(float* p, int n) {
    int i = blockIdx.x * 256 + threadIdx.x;
    if (i < n) p[i] = 0.0f;
}

// ---------------- GEMM: [nrows,128] bf16 @ packed [128,128] -> f32 --------
// 8 waves/block; wave w owns column tile w; one 16-row A tile staged in LDS.
__global__ void __launch_bounds__(256)
k_gemm128(const unsigned short* __restrict__ A,
          const unsigned short* __restrict__ Wp,
          float* __restrict__ Out, int nrows) {
    __shared__ unsigned short As[16 * 128];     // 4 KB bf16 tile
    int tid = threadIdx.x, lane = tid & 31, wave = tid >> 5;
    int m0 = blockIdx.x * 16;
    if (m0 >= nrows) return;                    // block-uniform
    ((u32x4*)As)[tid] = ((const u32x4*)(A + (size_t)m0 * 128))[tid]; // 256*16B = 4KB
    __syncthreads();

    v8f acc = {};
    int mL = lane & 15, kg = (lane >> 4) * 8;
    const u32x4* wp4 = (const u32x4*)Wp;
#pragma unroll
    for (int kt = 0; kt < 4; ++kt) {
        AF a, b;
        const u32x4* pa = (const u32x4*)(As + mL * 128 + kt * 32 + kg);
        a.q[0] = pa[0];                         // K = k0+kg .. +7
        a.q[1] = pa[2];                         // K = k0+kg+16 .. +23
        int idx = ((kt * 8 + wave) * 32 + lane) * 2;
        b.q[0] = wp4[idx];
        b.q[1] = wp4[idx + 1];
        acc = __builtin_amdgcn_wmma_f32_16x16x32_bf16(
            false, a.v, false, b.v, (short)0, acc, false, false);
    }
    int n  = wave * 16 + mL;
    int mb = m0 + (lane >> 4) * 8;
#pragma unroll
    for (int r = 0; r < 8; ++r)
        Out[(size_t)(mb + r) * 128 + n] = acc[r];
}

// ---------------- scatter-add aggregation (E edges, 4 chans/thread) -------
__global__ void k_aggregate(const int* __restrict__ ei,
                            const float* __restrict__ dinv,
                            const float* __restrict__ t,
                            float* __restrict__ z, int E) {
    int tid = blockIdx.x * 256 + threadIdx.x;
    int e = tid >> 5;
    if (e >= E) return;
    int c4 = (tid & 31) * 4;
    int r = ei[e], c = ei[E + e];
    float norm = dinv[r] * dinv[c];
    f32x4 v = *(const f32x4*)(t + (size_t)r * 128 + c4);
    float* zp = z + (size_t)c * 128 + c4;
    atomicAdd(zp + 0, v.x * norm);
    atomicAdd(zp + 1, v.y * norm);
    atomicAdd(zp + 2, v.z * norm);
    atomicAdd(zp + 3, v.w * norm);
}

// ---------------- self-loop + bias (+ReLU) + bf16 convert ----------------
__global__ void k_self_bias_act(const float* __restrict__ t,
                                const float* __restrict__ dinv,
                                const float* __restrict__ b,
                                float* __restrict__ z,
                                unsigned short* __restrict__ zb,
                                int n, int do_relu) {
    int tid = blockIdx.x * 256 + threadIdx.x;
    int i = tid >> 5;
    if (i >= n) return;
    int c = (tid & 31) * 4;
    float di = dinv[i], s = di * di;
    f32x4 tv = *(const f32x4*)(t + (size_t)i * 128 + c);
    f32x4 zv = *(const f32x4*)(z + (size_t)i * 128 + c);
    f32x4 bv = *(const f32x4*)(b + c);
    f32x4 o = zv + tv * s + bv;
    if (do_relu) {
        o.x = fmaxf(o.x, 0.f); o.y = fmaxf(o.y, 0.f);
        o.z = fmaxf(o.z, 0.f); o.w = fmaxf(o.w, 0.f);
    }
    *(f32x4*)(z + (size_t)i * 128 + c) = o;
    u32x2 w;
    w.x = (unsigned)f2bf(o.x) | ((unsigned)f2bf(o.y) << 16);
    w.y = (unsigned)f2bf(o.z) | ((unsigned)f2bf(o.w) << 16);
    *(u32x2*)(zb + (size_t)i * 128 + c) = w;
}

// ---------------- fused decoder: 16 edges / wave -------------------------
// ef[e] = concat(zb[row[e]], zb[col[e]]) (256);  relu(ef@Wd1+bd1)@Wd2+bd2
__global__ void __launch_bounds__(256)
k_decoder(const int* __restrict__ ei, const unsigned short* __restrict__ zb,
          const unsigned short* __restrict__ wd1p,
          const float* __restrict__ wd2, const float* __restrict__ bd1,
          const float* __restrict__ bd2, float* __restrict__ out, int E) {
    int lane = threadIdx.x & 31, wave = threadIdx.x >> 5;
    int tile = blockIdx.x * 8 + wave;
    if (tile * 16 >= E) return;                 // wave-uniform
    int e  = tile * 16 + (lane & 15);
    int nr = ei[e], nc = ei[E + e];

    v8f acc[4] = {};
    const u32x4* wp4 = (const u32x4*)wd1p;
    int kg = (lane >> 4) * 8;
#pragma unroll
    for (int kt = 0; kt < 8; ++kt) {
        int f0   = kt * 32 + kg;
        int node = (f0 < 128) ? nr : nc;        // 128 boundary is 32-aligned
        int g    = f0 & 127;
        const u32x4* pa = (const u32x4*)(zb + (size_t)node * 128 + g);
        AF a;
        a.q[0] = pa[0];
        a.q[1] = pa[2];
#pragma unroll
        for (int t = 0; t < 4; ++t) {
            int idx = ((kt * 4 + t) * 32 + lane) * 2;
            AF b;
            b.q[0] = wp4[idx];
            b.q[1] = wp4[idx + 1];
            acc[t] = __builtin_amdgcn_wmma_f32_16x16x32_bf16(
                false, a.v, false, b.v, (short)0, acc[t], false, false);
        }
    }
    // epilogue: per-lane col = t*16+(lane&15); rows r + 8*(lane>>4)
    int ln = lane & 15;
    float w2v[4], b1v[4];
#pragma unroll
    for (int t = 0; t < 4; ++t) { w2v[t] = wd2[t * 16 + ln]; b1v[t] = bd1[t * 16 + ln]; }
    float bd2v = bd2[0];
    float res[8];
#pragma unroll
    for (int r = 0; r < 8; ++r) {
        float p = fmaxf(acc[0][r] + b1v[0], 0.f) * w2v[0]
                + fmaxf(acc[1][r] + b1v[1], 0.f) * w2v[1]
                + fmaxf(acc[2][r] + b1v[2], 0.f) * w2v[2]
                + fmaxf(acc[3][r] + b1v[3], 0.f) * w2v[3];
        p += __shfl_xor(p, 1, 32);              // masks <16: stays in half
        p += __shfl_xor(p, 2, 32);
        p += __shfl_xor(p, 4, 32);
        p += __shfl_xor(p, 8, 32);
        res[r] = p;
    }
    if (ln == 0) {
        int base = tile * 16 + (lane >> 4) * 8;
#pragma unroll
        for (int r = 0; r < 8; ++r) out[base + r] = res[r] + bd2v;
    }
}

// ---------------------------------------------------------------------------
extern "C" void kernel_launch(void* const* d_in, const int* in_sizes, int n_in,
                              void* d_out, int out_size, void* d_ws, size_t ws_size,
                              hipStream_t stream) {
    const float* x   = (const float*)d_in[0];
    const int*   ei  = (const int*)  d_in[1];
    const float* W1  = (const float*)d_in[2];
    const float* b1  = (const float*)d_in[3];
    const float* W2  = (const float*)d_in[4];
    const float* b2  = (const float*)d_in[5];
    const float* Wd1 = (const float*)d_in[6];
    const float* bd1 = (const float*)d_in[7];
    const float* Wd2 = (const float*)d_in[8];
    const float* bd2 = (const float*)d_in[9];
    float* out = (float*)d_out;

    const int N = in_sizes[0] / 128;   // 50000
    const int E = in_sizes[1] / 2;     // 800000

    // workspace carve (~64.3 MB), 256B-aligned slices
    char* ws = (char*)d_ws;
    size_t off = 0;
    auto carve = [&](size_t bytes) {
        void* p = ws + off;
        off = (off + bytes + 255) & ~(size_t)255;
        return p;
    };
    float*          dinv = (float*)         carve((size_t)N * 4);
    unsigned short* xb   = (unsigned short*)carve((size_t)N * 128 * 2); // reused as zb
    unsigned short* w1p  = (unsigned short*)carve(128 * 128 * 2);
    unsigned short* w2p  = (unsigned short*)carve(128 * 128 * 2);
    unsigned short* wd1p = (unsigned short*)carve(256 * 64 * 2);
    float*          tbuf = (float*)         carve((size_t)N * 128 * 4);
    float*          zbuf = (float*)         carve((size_t)N * 128 * 4);
    unsigned short* zb   = xb;              // x only needed for GEMM1

    // 1) degree + normalization
    k_init_deg<<<(N + 255) / 256, 256, 0, stream>>>(dinv, N);
    k_count_deg<<<(E + 255) / 256, 256, 0, stream>>>(ei, dinv, E);
    k_rsqrt<<<(N + 255) / 256, 256, 0, stream>>>(dinv, N);

    // 2) precision conversion / weight packing
    k_f32_to_bf16_x4<<<(N * 32 + 255) / 256, 256, 0, stream>>>(x, xb, N * 32);
    k_pack_w<<<(128 * 128 + 255) / 256, 256, 0, stream>>>(W1, w1p, 128, 128);
    k_pack_w<<<(128 * 128 + 255) / 256, 256, 0, stream>>>(W2, w2p, 128, 128);
    k_pack_w<<<(256 * 64 + 255) / 256, 256, 0, stream>>>(Wd1, wd1p, 256, 64);

    // 3) layer 1: GEMM -> scatter -> self-loop+bias+ReLU (+bf16)
    k_gemm128<<<(N + 15) / 16, 256, 0, stream>>>(xb, w1p, tbuf, N);
    k_zero<<<(N * 128 + 255) / 256, 256, 0, stream>>>(zbuf, N * 128);
    k_aggregate<<<(E * 32 + 255) / 256, 256, 0, stream>>>(ei, dinv, tbuf, zbuf, E);
    k_self_bias_act<<<(N * 32 + 255) / 256, 256, 0, stream>>>(tbuf, dinv, b1, zbuf, zb, N, 1);

    // 4) layer 2 (no ReLU)
    k_gemm128<<<(N + 15) / 16, 256, 0, stream>>>(zb, w2p, tbuf, N);
    k_zero<<<(N * 128 + 255) / 256, 256, 0, stream>>>(zbuf, N * 128);
    k_aggregate<<<(E * 32 + 255) / 256, 256, 0, stream>>>(ei, dinv, tbuf, zbuf, E);
    k_self_bias_act<<<(N * 32 + 255) / 256, 256, 0, stream>>>(tbuf, dinv, b2, zbuf, zb, N, 0);

    // 5) fused edge decoder
    k_decoder<<<(E / 16 + 7) / 8, 256, 0, stream>>>(ei, zb, wd1p, Wd2, bd1, bd2, out, E);
}
